// Qwen3MoeAttention_9053791060132
// MI455X (gfx1250) — matrix operand order repaired and checked by reading, not compiled, not run on hardware
//
#include <hip/hip_runtime.h>
#include <cstdint>

// ---------------------------------------------------------------------------
// Qwen3 attention block for MI455X (gfx1250, wave32, WMMA + TDM).
// Precision: bf16 operands + fp32 accumulate via v_wmma_f32_16x16x32_bf16.
// Softmax / RMSNorm / RoPE in fp32 VALU. GEMMs stage tiles in LDS via the
// Tensor Data Mover (tensor_load_to_lds + s_wait_tensorcnt) when available.
// ---------------------------------------------------------------------------

typedef unsigned short u16;
typedef unsigned int   u32;
typedef __attribute__((ext_vector_type(16))) __bf16 v16bf;
typedef __attribute__((ext_vector_type(8)))  float  v8f;
typedef __attribute__((ext_vector_type(8)))  u32    v8u;
typedef __attribute__((ext_vector_type(4)))  u32    v4u;
typedef __attribute__((ext_vector_type(8)))  int    v8i;
typedef __attribute__((ext_vector_type(4)))  int    v4i;

#define WMMA_BF16(A_, B_, C_) \
  __builtin_amdgcn_wmma_f32_16x16x32_bf16(false, (A_), false, (B_), (short)0, (C_), false, false)

__device__ __forceinline__ u16 f2bf(float x) {
  u32 u = __float_as_uint(x);
  u += 0x7FFFu + ((u >> 16) & 1u);   // round-to-nearest-even
  return (u16)(u >> 16);
}

// Build a fragment (8 dwords) from two 16-byte chunks at p and p+second_off.
__device__ __forceinline__ v16bf frag_2x16(const u16* p, int second_off) {
  uint4 a = *(const uint4*)p;
  uint4 b = *(const uint4*)(p + second_off);
  v8u f;
  f[0] = a.x; f[1] = a.y; f[2] = a.z; f[3] = a.w;
  f[4] = b.x; f[5] = b.y; f[6] = b.z; f[7] = b.w;
  return __builtin_bit_cast(v16bf, f);
}

// A fragment: 16(M) x 32(K) bf16 from row-major bf16 [rows][lda].
// ISA layout: lane = m + 16*((k>>3)&1); vgpr = 4*(k>>4) + ((k&7)>>1); half = k&1.
__device__ __forceinline__ v16bf load_frag_a(const u16* __restrict__ A, int lda,
                                             int row0, int kc, int lane) {
  int m = lane & 15, khi = lane >> 4;
  return frag_2x16(A + (size_t)(row0 + m) * lda + kc + 8 * khi, 16);
}

// B fragment: 32(K) x 16(N) with B(k,n) = Bt[n0+n][kc+k], Bt row-major [N][ldb].
// ISA layout: lane = n + 16*((k>>3)&1); vgpr = 4*(k>>4) + ((k&7)>>1); half = k&1.
__device__ __forceinline__ v16bf load_frag_b(const u16* __restrict__ Bt, int ldb,
                                             int n0, int kc, int lane) {
  int n = lane & 15, kg = lane >> 4;
  return frag_2x16(Bt + (size_t)(n0 + n) * ldb + kc + 8 * kg, 16);
}

// ---------------------------------------------------------------------------
// TDM helper: DMA a 128-row x 32-element bf16 tile (row-major, row stride
// lda elements) from global into LDS (contiguous [128][32]).
// D# per CDNA5 ISA 8.3-8.6: group0 {count=1, lds_addr, global_addr, type=2},
// group1 {data_size=1(2B), tensor_dim0=lda, tensor_dim1=1<<20, tile=32x128,
// dim0_stride=lda}. Groups 2/3 zero (2D tile).
// ---------------------------------------------------------------------------
#if defined(__HIP_DEVICE_COMPILE__) && __has_builtin(__builtin_amdgcn_tensor_load_to_lds)
#define ATHENA_HAVE_TDM 1
__device__ __forceinline__ void tdm_tile_load(unsigned lds_byte_off,
                                              const u16* gsrc, int lda_elems) {
  unsigned long long ga = (unsigned long long)(uintptr_t)gsrc;
  v4u g0 = {1u,                                    // count = 1 valid descriptor
            lds_byte_off,                          // lds_addr (bytes)
            (u32)ga,                               // global_addr[31:0]
            ((u32)(ga >> 32) & 0x01FFFFFFu) | (2u << 30)}; // addr[56:32] | type=2
  u32 lda = (u32)lda_elems;
  v8i g1 = {(int)(1u << 16),                       // data_size = 1 -> 2 bytes
            (int)((lda & 0xFFFFu) << 16),          // tensor_dim0 lo16 @ [63:48]
            (int)((lda >> 16) | (0u << 16)),       // dim0 hi16 | dim1 lo16 (dim1=1<<20)
            (int)(16u | (32u << 16)),              // dim1 hi16 (=16 -> 1<<20) | tile_dim0=32
            (int)128,                              // tile_dim1=128, tile_dim2=0
            (int)lda,                              // tensor_dim0_stride lo32
            0, 0};
  v4i z4 = {0, 0, 0, 0};
#if __has_include(<hip/amd_detail/amd_gfx1250_TDM.h>)
  v8i z8 = {0, 0, 0, 0, 0, 0, 0, 0};
  __builtin_amdgcn_tensor_load_to_lds(g0, g1, z4, z4, z8, 0);
#else
  __builtin_amdgcn_tensor_load_to_lds(g0, g1, z4, z4, 0);
#endif
}
#endif

// ---------------------------------------------------------------------------
// Converters
// ---------------------------------------------------------------------------
__global__ void cvt_f32_bf16(const float* __restrict__ in, u16* __restrict__ out, int n) {
  int i = blockIdx.x * blockDim.x + threadIdx.x;
  if (i < n) out[i] = f2bf(in[i]);
}

// in: fp32 [R][C] row-major  ->  out: bf16 [C][R] row-major (transposed)
__global__ void cvt_transpose_bf16(const float* __restrict__ in, u16* __restrict__ out,
                                   int R, int C) {
  int i = blockIdx.x * blockDim.x + threadIdx.x;
  if (i < R * C) {
    int r = i / C, c = i - r * C;
    out[(size_t)c * R + r] = f2bf(in[i]);
  }
}

// ---------------------------------------------------------------------------
// Tiled GEMM: C[M][N] (fp32) = A[M][K] (bf16 row-major) x Bt[N][K]^T (bf16).
// Workgroup = 256 threads (8 waves) computes a 128(M) x 128(N) tile.
// Wave w: 32x64 subtile at (m_off=(w&3)*32, n_off=(w>>2)*64): 8 accumulators,
// 8 WMMAs per 32-deep K step from LDS-staged 128x32 A and B tiles.
// Staging: TDM double-buffer (wave 0 issues tensor_load_to_lds, waits
// tensorcnt, barrier publishes) or cooperative-copy fallback.
// grid = (N/128, M/128)
// ---------------------------------------------------------------------------
__global__ __launch_bounds__(256) void gemm_bf16_tiled(const u16* __restrict__ A,
                                                       const u16* __restrict__ Bt,
                                                       float* __restrict__ C,
                                                       int M, int N, int K) {
  __shared__ u16 tA[2][128][32];
  __shared__ u16 tB[2][128][32];
  int lane = threadIdx.x & 31;
  int wave = threadIdx.x >> 5;
  int m0 = blockIdx.y * 128;
  int n0 = blockIdx.x * 128;
  int m_off = (wave & 3) * 32;
  int n_off = (wave >> 2) * 64;
  int nl = lane & 15, hi = lane >> 4;

  v8f z = {0.f, 0.f, 0.f, 0.f, 0.f, 0.f, 0.f, 0.f};
  v8f acc[2][4];
#pragma unroll
  for (int sm = 0; sm < 2; sm++)
#pragma unroll
    for (int t = 0; t < 4; t++) acc[sm][t] = z;

  int niter = K >> 5;

#ifdef ATHENA_HAVE_TDM
  // ---- TDM double-buffered staging ----
  unsigned ldsA = (unsigned)(size_t)(&tA[0][0][0]);   // addr[31:0] == LDS offset
  unsigned ldsB = (unsigned)(size_t)(&tB[0][0][0]);
  const unsigned bufBytes = 128 * 32 * 2;

  if (wave == 0) {
    tdm_tile_load(ldsA, A + (size_t)m0 * K, K);
    tdm_tile_load(ldsB, Bt + (size_t)n0 * K, K);
  }
  for (int it = 0; it < niter; ++it) {
    int cur = it & 1;
    if (wave == 0) __builtin_amdgcn_s_wait_tensorcnt(0);
    __syncthreads();                       // tiles in buf[cur] visible to all
    if (wave == 0 && it + 1 < niter) {     // prefetch buf[cur^1]
      int kc = (it + 1) * 32;
      unsigned off = (unsigned)(cur ^ 1) * bufBytes;
      tdm_tile_load(ldsA + off, A + (size_t)m0 * K + kc, K);
      tdm_tile_load(ldsB + off, Bt + (size_t)n0 * K + kc, K);
    }
    v16bf fa0 = frag_2x16(&tA[cur][m_off + nl][8 * hi], 16);
    v16bf fa1 = frag_2x16(&tA[cur][m_off + 16 + nl][8 * hi], 16);
#pragma unroll
    for (int t = 0; t < 4; ++t) {
      v16bf fb = frag_2x16(&tB[cur][n_off + 16 * t + nl][8 * hi], 16);
      acc[0][t] = WMMA_BF16(fa0, fb, acc[0][t]);
      acc[1][t] = WMMA_BF16(fa1, fb, acc[1][t]);
    }
    // No bottom barrier: buf[cur] is only re-issued after the *next* top
    // barrier, which every wave reaches only after finishing this compute.
  }
#else
  // ---- cooperative-copy staging fallback (single buffer) ----
  for (int it = 0; it < niter; ++it) {
    int kc = it * 32;
    __syncthreads();                       // previous reads done before overwrite
    for (int j = threadIdx.x; j < 512; j += 256) {
      int r = j >> 2, c = (j & 3) * 8;
      *(uint4*)&tA[0][r][c] = *(const uint4*)(A + (size_t)(m0 + r) * K + kc + c);
      *(uint4*)&tB[0][r][c] = *(const uint4*)(Bt + (size_t)(n0 + r) * K + kc + c);
    }
    __syncthreads();
    v16bf fa0 = frag_2x16(&tA[0][m_off + nl][8 * hi], 16);
    v16bf fa1 = frag_2x16(&tA[0][m_off + 16 + nl][8 * hi], 16);
#pragma unroll
    for (int t = 0; t < 4; ++t) {
      v16bf fb = frag_2x16(&tB[0][n_off + 16 * t + nl][8 * hi], 16);
      acc[0][t] = WMMA_BF16(fa0, fb, acc[0][t]);
      acc[1][t] = WMMA_BF16(fa1, fb, acc[1][t]);
    }
  }
#endif

#pragma unroll
  for (int sm = 0; sm < 2; sm++)
#pragma unroll
    for (int t = 0; t < 4; t++)
#pragma unroll
      for (int v = 0; v < 8; v++) {
        size_t r = (size_t)(m0 + m_off + 16 * sm + v + 8 * hi) * N
                 + n0 + n_off + 16 * t + nl;
        C[r] = acc[sm][t][v];
      }
}

// ---------------------------------------------------------------------------
// RMSNorm + NeoX RoPE + layout reorg.
// grid = (S=2048, 40): y in [0,32) -> Q head, [32,36) -> K head, [36,40) -> V head.
// block = 128 threads (one per head-dim element).
// Q gets the 1/sqrt(128) attention scale folded in (linear, commutes with RoPE).
// ---------------------------------------------------------------------------
__global__ __launch_bounds__(128) void normrope(const float* __restrict__ qkv,
                                                const int* __restrict__ pos,
                                                const float* __restrict__ qw,
                                                const float* __restrict__ kw,
                                                u16* __restrict__ qb,   // [S][32][128]
                                                u16* __restrict__ kb,   // [S][4][128]
                                                u16* __restrict__ vt) { // [4][128][S]
  const int S = 2048;
  int s = blockIdx.x;
  int slot = blockIdx.y;
  int d = threadIdx.x;
  const float* row = qkv + (size_t)s * 5120;

  if (slot >= 36) {                    // V: straight convert, transposed store
    int h = slot - 36;
    vt[((size_t)h * 128 + d) * S + s] = f2bf(row[4608 + h * 128 + d]);
    return;
  }

  __shared__ float xs[128];
  __shared__ float wsum[4];
  bool isq = (slot < 32);
  int h = isq ? slot : slot - 32;
  int off = isq ? h * 128 : 4096 + h * 128;
  float x = row[off + d];

  float ss = x * x;
#pragma unroll
  for (int o = 16; o > 0; o >>= 1) ss += __shfl_xor(ss, o, 32);
  int wv = threadIdx.x >> 5;
  if ((threadIdx.x & 31) == 0) wsum[wv] = ss;
  __syncthreads();
  float tot = wsum[0] + wsum[1] + wsum[2] + wsum[3];
  float rs = 1.0f / sqrtf(tot * (1.0f / 128.0f) + 1e-6f);

  float w = isq ? qw[d] : kw[d];
  float xn = x * rs * w;
  if (isq) xn *= 0.08838834764831845f;     // 128^-0.5 folded into Q
  xs[d] = xn;
  __syncthreads();

  int fi = d & 63;
  // inv_freq = THETA^(-2*fi/128), ln(1e6) = 13.815510557964274
  float inv = expf(-13.815510557964274f * (float)(2 * fi) * (1.0f / 128.0f));
  float ang = (float)pos[s] * inv;
  float c = cosf(ang), sn = sinf(ang);
  float other = xs[d ^ 64];
  float outv = (d < 64) ? (xn * c - other * sn) : (xn * c + other * sn);

  if (isq) qb[((size_t)s * 32 + h) * 128 + d] = f2bf(outv);
  else     kb[((size_t)s * 4  + h) * 128 + d] = f2bf(outv);
}

// ---------------------------------------------------------------------------
// Flash attention (causal, GQA rep=8). One wave per (head, 16-query tile).
// block = 128 threads (4 waves); grid = (32 heads, 2048/64).
// Per 32-key block: QK^T via 2x(4 WMMA), online softmax (fp32), P staged
// through LDS (C-layout -> A-layout), PV via 8 WMMA into 16x128 accumulators.
// ---------------------------------------------------------------------------
__global__ __launch_bounds__(128) void attn_kernel(const u16* __restrict__ Qb,
                                                   const u16* __restrict__ Kb,
                                                   const u16* __restrict__ Vtb,
                                                   u16* __restrict__ Ob) { // [S][4096] bf16
  const int S = 2048;
  int lane = threadIdx.x & 31;
  int wave = threadIdx.x >> 5;
  int h = blockIdx.x;
  int q0 = (blockIdx.y * 4 + wave) * 16;
  int kvh = h >> 3;
  int nl = lane & 15, hi = lane >> 4;

  __shared__ float Plds[4][16][32];

  const u16* Qbase = Qb + h * 128;                  // row stride 4096
  const u16* Kbase = Kb + kvh * 128;                // row stride 512
  const u16* Vbase = Vtb + (size_t)kvh * 128 * S;   // row stride S

  v16bf qf[4];
#pragma unroll
  for (int dc = 0; dc < 4; dc++) qf[dc] = load_frag_a(Qbase, 4096, q0, dc * 32, lane);

  v8f z = {0.f, 0.f, 0.f, 0.f, 0.f, 0.f, 0.f, 0.f};
  v8f acc[8];
#pragma unroll
  for (int t = 0; t < 8; t++) acc[t] = z;
  float mrow[8], lrow[8];
#pragma unroll
  for (int v = 0; v < 8; v++) { mrow[v] = -INFINITY; lrow[v] = 0.f; }

  int nkb = (q0 + 47) >> 5;   // enough 32-key blocks to cover keys <= q0+15
  for (int kbk = 0; kbk < nkb; kbk++) {
    float p[2][8];
    float bmax[8];
#pragma unroll
    for (int v = 0; v < 8; v++) bmax[v] = -INFINITY;

#pragma unroll
    for (int sub = 0; sub < 2; sub++) {
      int k0 = kbk * 32 + sub * 16;
      v8f sacc = z;
#pragma unroll
      for (int dc = 0; dc < 4; dc++) {
        v16bf kf = load_frag_b(Kbase, 512, k0, dc * 32, lane);
        sacc = WMMA_BF16(qf[dc], kf, sacc);
      }
      int kg = k0 + nl;
#pragma unroll
      for (int v = 0; v < 8; v++) {
        float sv = sacc[v];
        if (kg > q0 + v + 8 * hi) sv = -INFINITY;   // causal mask
        p[sub][v] = sv;
        bmax[v] = fmaxf(bmax[v], sv);
      }
    }

    // row-max across the 16 lanes of each half
#pragma unroll
    for (int o = 1; o < 16; o <<= 1)
#pragma unroll
      for (int v = 0; v < 8; v++) bmax[v] = fmaxf(bmax[v], __shfl_xor(bmax[v], o, 32));

    float sc[8];
#pragma unroll
    for (int v = 0; v < 8; v++) {
      float mn = fmaxf(mrow[v], bmax[v]);
      sc[v] = __expf(mrow[v] - mn);
      mrow[v] = mn;
    }

    float psum[8];
#pragma unroll
    for (int v = 0; v < 8; v++) psum[v] = 0.f;
#pragma unroll
    for (int sub = 0; sub < 2; sub++)
#pragma unroll
      for (int v = 0; v < 8; v++) {
        float pv = __expf(p[sub][v] - mrow[v]);
        psum[v] += pv;
        Plds[wave][v + 8 * hi][sub * 16 + nl] = pv;   // C-layout -> LDS
      }
#pragma unroll
    for (int o = 1; o < 16; o <<= 1)
#pragma unroll
      for (int v = 0; v < 8; v++) psum[v] += __shfl_xor(psum[v], o, 32);
#pragma unroll
    for (int v = 0; v < 8; v++) lrow[v] = lrow[v] * sc[v] + psum[v];

    // rescale accumulators
#pragma unroll
    for (int t = 0; t < 8; t++)
#pragma unroll
      for (int v = 0; v < 8; v++) acc[t][v] *= sc[v];

    // rebuild P as bf16 A-fragment (16 x 32) from LDS
    v8u pf;
#pragma unroll
    for (int v = 0; v < 8; v++) {
      int k = 16 * (v >> 2) + 8 * hi + 2 * (v & 3);
      float a0 = Plds[wave][nl][k];
      float a1 = Plds[wave][nl][k + 1];
      pf[v] = (u32)f2bf(a0) | ((u32)f2bf(a1) << 16);
    }
    v16bf pa = __builtin_bit_cast(v16bf, pf);

    // P x V over the 8 output 16-dim tiles
#pragma unroll
    for (int t = 0; t < 8; t++) {
      v16bf vf = load_frag_b(Vbase, S, t * 16, kbk * 32, lane);
      acc[t] = WMMA_BF16(pa, vf, acc[t]);
    }
  }

  // epilogue: divide by row sums, store bf16 [S][32*128]
  float il[8];
#pragma unroll
  for (int v = 0; v < 8; v++) il[v] = 1.0f / lrow[v];
#pragma unroll
  for (int t = 0; t < 8; t++)
#pragma unroll
    for (int v = 0; v < 8; v++) {
      size_t idx = (size_t)(q0 + v + 8 * hi) * 4096 + h * 128 + t * 16 + nl;
      Ob[idx] = f2bf(acc[t][v] * il[v]);
    }
}

// ---------------------------------------------------------------------------
// Orchestration
// ---------------------------------------------------------------------------
extern "C" void kernel_launch(void* const* d_in, const int* in_sizes, int n_in,
                              void* d_out, int out_size, void* d_ws, size_t ws_size,
                              hipStream_t stream) {
  (void)in_sizes; (void)n_in; (void)out_size; (void)ws_size;
  const float* hidden    = (const float*)d_in[0];   // [1,2048,2048]
  const int*   positions = (const int*)d_in[1];     // [2048]
  const float* w_qkv     = (const float*)d_in[2];   // [2048,5120]
  const float* w_o       = (const float*)d_in[3];   // [4096,2048]
  const float* qw        = (const float*)d_in[4];   // [128]
  const float* kw        = (const float*)d_in[5];   // [128]
  float* out = (float*)d_out;                       // [2048,2048]

  char* ws = (char*)d_ws;
  u16*   hb    = (u16*)(ws + 0);           //  8.0 MiB  hidden bf16 [2048][2048]
  u16*   wqkvT = (u16*)(ws + 8388608);     // 20.0 MiB  w_qkv^T bf16 [5120][2048]
  u16*   woT   = (u16*)(ws + 29360128);    // 16.0 MiB  w_o^T bf16 [2048][4096]
  float* qkv   = (float*)(ws + 46137344);  // 40.0 MiB  qkv fp32 [2048][5120]
  u16*   qb    = (u16*)(ws + 88080384);    // 16.0 MiB  q bf16 [2048][32][128]
  u16*   kb    = (u16*)(ws + 104857600);   //  2.0 MiB  k bf16 [2048][4][128]
  u16*   vt    = (u16*)(ws + 106954752);   //  2.0 MiB  v^T bf16 [4][128][2048]
  u16*   attn  = (u16*)(ws + 109051904);   // 16.0 MiB  attn bf16 [2048][4096]

  cvt_f32_bf16<<<(2048 * 2048 + 255) / 256, 256, 0, stream>>>(hidden, hb, 2048 * 2048);
  cvt_transpose_bf16<<<(2048 * 5120 + 255) / 256, 256, 0, stream>>>(w_qkv, wqkvT, 2048, 5120);
  cvt_transpose_bf16<<<(4096 * 2048 + 255) / 256, 256, 0, stream>>>(w_o, woT, 4096, 2048);

  // QKV projection: [2048,2048] x [2048,5120]
  gemm_bf16_tiled<<<dim3(5120 / 128, 2048 / 128), 256, 0, stream>>>(
      hb, wqkvT, qkv, 2048, 5120, 2048);

  normrope<<<dim3(2048, 40), 128, 0, stream>>>(qkv, positions, qw, kw, qb, kb, vt);

  attn_kernel<<<dim3(32, 32), 128, 0, stream>>>(qb, kb, vt, attn);

  // Output projection: [2048,4096] x [4096,2048]
  gemm_bf16_tiled<<<dim3(2048 / 128, 2048 / 128), 256, 0, stream>>>(
      attn, woT, out, 2048, 2048, 4096);
}
